// SparseAttention_687194768163
// MI455X (gfx1250) — matrix operand order repaired and checked by reading, not compiled, not run on hardware
//
#include <hip/hip_runtime.h>
#include <math.h>

// ---------------------------------------------------------------------------
// Problem constants (from reference): B=2, L=2048, C=512, H=8, D=64
// BLKQ=128 -> nq=16, BLKK=64 -> nk=32, scale = D^-0.5 = 0.125
// ---------------------------------------------------------------------------
#define BB    2
#define LL    2048
#define CC    512
#define HH    8
#define DD    64
#define NQ    16
#define NK    32
#define SCALE 0.125f

typedef __attribute__((ext_vector_type(16))) __bf16 v16bf;
typedef __attribute__((ext_vector_type(8)))  float  v8f;
typedef __attribute__((ext_vector_type(4)))  int    v4i;

union BF16x16 { v16bf v; unsigned short u[16]; };
union F32x8   { v8f   v; float          f[8];  };

#define AS1 __attribute__((address_space(1)))
#define AS3 __attribute__((address_space(3)))

__device__ __forceinline__ unsigned short f2bf(float f) {
    unsigned int x = __float_as_uint(f);
    unsigned int r = (x + 0x7FFFu + ((x >> 16) & 1u)) >> 16;   // RNE
    return (unsigned short)r;
}

// A-matrix (16x32 bf16) element -> K index inside the 32-chunk
__device__ __forceinline__ int a_kidx(int e, int half) {
    int vg = e >> 1, sub = e & 1;
    return ((vg < 4) ? (2 * vg) : (16 + 2 * (vg - 4))) + sub + (half ? 8 : 0);
}

// Wait for all outstanding async (global<->LDS) transfers.
__device__ __forceinline__ void wait_async0() {
#if __has_builtin(__builtin_amdgcn_s_wait_asynccnt)
    __builtin_amdgcn_s_wait_asynccnt(0);
#else
    asm volatile("s_wait_asynccnt 0x0" ::: "memory");
#endif
}

// ---------------------------------------------------------------------------
// Kernel 1: QKV = x @ Wqkv + bqkv ; scatter to q/k/v in [B,H,L,D] layout.
// f32 copies of q,k kept for block statistics, bf16 copies for WMMA attention.
// Tiles: 256 (M) x 96 (N) 16x16 tiles, 1 tile per wave, 8 waves per block.
// ---------------------------------------------------------------------------
__global__ void k_qkv_gemm(const float* __restrict__ x,
                           const float* __restrict__ Wqkv,
                           const float* __restrict__ bqkv,
                           float* __restrict__ qf, float* __restrict__ kf,
                           unsigned short* __restrict__ qbf,
                           unsigned short* __restrict__ kbf,
                           unsigned short* __restrict__ vbf) {
    const int gw   = blockIdx.x * 8 + (threadIdx.x >> 5);
    const int lane = threadIdx.x & 31;
    const int half = lane >> 4, r = lane & 15;
    const int tM = gw / 96, tN = gw % 96;

    F32x8 c;
#pragma unroll
    for (int i = 0; i < 8; ++i) c.f[i] = 0.0f;

    for (int k = 0; k < CC; k += 32) {
        if (k + 32 < CC) {  // prefetch next weight slab -> global_prefetch_b8
            __builtin_prefetch(&Wqkv[(size_t)(k + 32 + 16 * half) * (3 * CC) + tN * 16 + r], 0, 1);
        }
        BF16x16 a, b;
#pragma unroll
        for (int e = 0; e < 16; ++e) {
            int ka = k + a_kidx(e, half);
            a.u[e] = f2bf(x[(size_t)(tM * 16 + r) * CC + ka]);
            int kb = k + a_kidx(e, half);
            b.u[e] = f2bf(Wqkv[(size_t)kb * (3 * CC) + tN * 16 + r]);
        }
        c.v = __builtin_amdgcn_wmma_f32_16x16x32_bf16(false, a.v, false, b.v,
                                                      (short)0, c.v, false, false);
    }

#pragma unroll
    for (int i = 0; i < 8; ++i) {
        int m = tM * 16 + i + 8 * half;
        int n = tN * 16 + r;
        float val = c.f[i] + bqkv[n];
        int b_  = m >> 11, l = m & (LL - 1);
        int s   = n >> 9;                 // 0=q 1=k 2=v
        int cc_ = n & (CC - 1);
        int h   = cc_ >> 6, d = cc_ & 63;
        size_t idx = (((size_t)b_ * HH + h) * LL + l) * DD + d;
        if (s == 0)      { qf[idx] = val; qbf[idx] = f2bf(val); }
        else if (s == 1) { kf[idx] = val; kbf[idx] = f2bf(val); }
        else             { vbf[idx] = f2bf(val); }
    }
}

// ---------------------------------------------------------------------------
// Kernel 2: block means + min cosine-to-mean. One block per (b*h, blockIdx).
// ---------------------------------------------------------------------------
__global__ void k_block_meta(const float* __restrict__ src, int blk, int nb,
                             float* __restrict__ meanOut,
                             float* __restrict__ simOut) {
    __shared__ float smean[DD];
    __shared__ float sred[128];
    const int tid = threadIdx.x;
    const int bh = blockIdx.x / nb, bi = blockIdx.x % nb;
    const size_t base = ((size_t)bh * LL + (size_t)bi * blk) * DD;

    if (tid < DD) {
        float s = 0.0f;
        for (int rr = 0; rr < blk; ++rr) s += src[base + (size_t)rr * DD + tid];
        float m = s / (float)blk;
        smean[tid] = m;
        meanOut[((size_t)bh * nb + bi) * DD + tid] = m;
    }
    __syncthreads();

    float mn2 = 0.0f;
#pragma unroll 8
    for (int d = 0; d < DD; ++d) mn2 += smean[d] * smean[d];
    float mnorm = sqrtf(mn2);

    float cosv = 1e30f;
    if (tid < blk) {
        float dot = 0.0f, n2 = 0.0f;
#pragma unroll 8
        for (int d = 0; d < DD; ++d) {
            float xv = src[base + (size_t)tid * DD + d];
            dot += xv * smean[d];
            n2  += xv * xv;
        }
        cosv = dot / ((sqrtf(n2) + 1e-6f) * (mnorm + 1e-6f));
    }
    sred[tid] = cosv;
    __syncthreads();
    for (int s = 64; s > 0; s >>= 1) {
        if (tid < s) sred[tid] = fminf(sred[tid], sred[tid + s]);
        __syncthreads();
    }
    if (tid == 0) simOut[(size_t)bh * nb + bi] = sred[0];
}

// ---------------------------------------------------------------------------
// Kernel 3: pooled softmax over block means + sort-free CDF keep mask.
// One wave per (b,h,qblock); lane j = K-block j.
// keep_i  <=>  sum{ p_j : (p_j > p_i) || (p_j == p_i && j < i) } < CDFTHRESHD
// ---------------------------------------------------------------------------
__global__ void k_pool_mask(const float* __restrict__ qm,
                            const float* __restrict__ km,
                            const float* __restrict__ qsim,
                            const float* __restrict__ ksim,
                            int* __restrict__ mask) {
    const int lane = threadIdx.x;            // 0..31 -> K block
    const int wg = blockIdx.x;               // (b*H+h)*NQ + qi
    const int bh = wg / NQ, qi = wg % NQ;

    const float* qv = qm + ((size_t)bh * NQ + qi) * DD;
    const float* kv = km + ((size_t)bh * NK + lane) * DD;
    float dot = 0.0f;
#pragma unroll 8
    for (int d = 0; d < DD; ++d) dot += qv[d] * kv[d];
    float s = dot * SCALE;

    float mx = s;
#pragma unroll
    for (int off = 16; off > 0; off >>= 1) mx = fmaxf(mx, __shfl_xor(mx, off));
    float p = __expf(s - mx);
    float sum = p;
#pragma unroll
    for (int off = 16; off > 0; off >>= 1) sum += __shfl_xor(sum, off);
    p /= sum;

    float before = 0.0f;
#pragma unroll
    for (int j = 0; j < NK; ++j) {
        float pj = __shfl(p, j);
        if ((pj > p) || (pj == p && j < lane)) before += pj;
    }
    bool keep = before < 0.98f;
    bool q_self = qsim[(size_t)bh * NQ + qi] > 0.6f;
    bool k_self = ksim[(size_t)bh * NK + lane] > 0.6f;
    mask[((size_t)bh * NQ + qi) * NK + lane] = (keep || !q_self || !k_self) ? 1 : 0;
}

// ---------------------------------------------------------------------------
// Kernel 4: block-sparse flash attention (bf16 WMMA, f32 online softmax).
// One workgroup = one (b,h,qblock=128 rows); 8 waves x 16 rows each.
// K/V 64x64 bf16 tiles staged to LDS via GLOBAL_LOAD_ASYNC_TO_LDS_B128
// (ASYNCcnt path); masked blocks skipped uniformly.
// ---------------------------------------------------------------------------
__global__ void k_sparse_attn(const unsigned short* __restrict__ qbf,
                              const unsigned short* __restrict__ kbf,
                              const unsigned short* __restrict__ vbf,
                              const int* __restrict__ mask,
                              unsigned short* __restrict__ obf) {
    __shared__ unsigned short lds_k[64 * 64];
    __shared__ unsigned short lds_v[64 * 64];
    __shared__ unsigned short lds_p[8][16 * 64];

    const int tid  = threadIdx.x;
    const int wave = tid >> 5, lane = tid & 31;
    const int half = lane >> 4, r = lane & 15;
    const int wg = blockIdx.x;
    const int b_ = wg / (HH * NQ);
    const int h  = (wg / NQ) % HH;
    const int qi = wg % NQ;
    const size_t bhBase = ((size_t)b_ * HH + h) * LL;
    const int qrow0 = qi * 128 + wave * 16;

    // Q A-fragments (16 rows x 64 d), two 16x32 chunks
    BF16x16 qa[2];
#pragma unroll
    for (int kk2 = 0; kk2 < 2; ++kk2)
#pragma unroll
        for (int e = 0; e < 16; ++e) {
            int d = kk2 * 32 + a_kidx(e, half);
            qa[kk2].u[e] = qbf[(bhBase + qrow0 + r) * DD + d];
        }

    F32x8 o[4];
#pragma unroll
    for (int nt = 0; nt < 4; ++nt)
#pragma unroll
        for (int i = 0; i < 8; ++i) o[nt].f[i] = 0.0f;
    float mrow[8], lrow[8];
#pragma unroll
    for (int i = 0; i < 8; ++i) { mrow[i] = -1e30f; lrow[i] = 0.0f; }

    for (int kb = 0; kb < NK; ++kb) {
        if (!mask[((size_t)(b_ * HH + h) * NQ + qi) * NK + kb]) continue;  // uniform
        __syncthreads();   // protect LDS reuse across iterations

        // cooperative stage of K/V 64x64 bf16 tiles (contiguous 8KB each)
        const uint4* gk = (const uint4*)(kbf + (bhBase + (size_t)kb * 64) * DD);
        const uint4* gv = (const uint4*)(vbf + (bhBase + (size_t)kb * 64) * DD);
        uint4* sk = (uint4*)lds_k;
        uint4* sv = (uint4*)lds_v;
#if __has_builtin(__builtin_amdgcn_global_load_async_to_lds_b128)
        // Async global->LDS DMA: no VGPR round-trip, tracked by ASYNCcnt.
#pragma unroll
        for (int t = 0; t < 2; ++t) {
            int i = tid + t * 256;
            __builtin_amdgcn_global_load_async_to_lds_b128(
                (AS1 v4i*)(gk + i), (AS3 v4i*)(sk + i), 0, 0);
            __builtin_amdgcn_global_load_async_to_lds_b128(
                (AS1 v4i*)(gv + i), (AS3 v4i*)(sv + i), 0, 0);
        }
        wait_async0();
#else
        for (int i = tid; i < 512; i += 256) { sk[i] = gk[i]; sv[i] = gv[i]; }
#endif
        __syncthreads();

        // S = q @ k^T  (4 N-tiles of 16 keys)
        F32x8 s[4];
#pragma unroll
        for (int nt = 0; nt < 4; ++nt) {
            F32x8 c;
#pragma unroll
            for (int i = 0; i < 8; ++i) c.f[i] = 0.0f;
#pragma unroll
            for (int kk2 = 0; kk2 < 2; ++kk2) {
                BF16x16 bfr;
#pragma unroll
                for (int e = 0; e < 16; ++e) {
                    int key = nt * 16 + r;
                    int d   = kk2 * 32 + a_kidx(e, half);
                    bfr.u[e] = lds_k[key * 64 + d];
                }
                c.v = __builtin_amdgcn_wmma_f32_16x16x32_bf16(
                        false, qa[kk2].v, false, bfr.v, (short)0, c.v, false, false);
            }
#pragma unroll
            for (int i = 0; i < 8; ++i) c.f[i] *= SCALE;
            s[nt] = c;
        }

        // online softmax per row (row = i + 8*half), columns across 16 lanes + 4 tiles
#pragma unroll
        for (int i = 0; i < 8; ++i) {
            float tm = s[0].f[i];
#pragma unroll
            for (int nt = 1; nt < 4; ++nt) tm = fmaxf(tm, s[nt].f[i]);
#pragma unroll
            for (int off = 8; off > 0; off >>= 1) tm = fmaxf(tm, __shfl_xor(tm, off));
            float mn = fmaxf(mrow[i], tm);
            float corr = __expf(mrow[i] - mn);
            float rs = 0.0f;
#pragma unroll
            for (int nt = 0; nt < 4; ++nt) {
                float pv = __expf(s[nt].f[i] - mn);
                s[nt].f[i] = pv;
                rs += pv;
            }
#pragma unroll
            for (int off = 8; off > 0; off >>= 1) rs += __shfl_xor(rs, off);
            lrow[i] = lrow[i] * corr + rs;
            mrow[i] = mn;
#pragma unroll
            for (int nt = 0; nt < 4; ++nt) o[nt].f[i] *= corr;
        }

        // stage P (16x64 bf16) through LDS to convert C-layout -> A-layout
#pragma unroll
        for (int nt = 0; nt < 4; ++nt)
#pragma unroll
            for (int i = 0; i < 8; ++i)
                lds_p[wave][(i + 8 * half) * 64 + nt * 16 + r] = f2bf(s[nt].f[i]);

        // O += P @ V
#pragma unroll
        for (int kk2 = 0; kk2 < 2; ++kk2) {
            BF16x16 pa;
#pragma unroll
            for (int e = 0; e < 16; ++e)
                pa.u[e] = lds_p[wave][r * 64 + kk2 * 32 + a_kidx(e, half)];
#pragma unroll
            for (int nt = 0; nt < 4; ++nt) {
                BF16x16 vb;
#pragma unroll
                for (int e = 0; e < 16; ++e) {
                    int key = kk2 * 32 + a_kidx(e, half);
                    vb.u[e] = lds_v[key * 64 + nt * 16 + r];
                }
                o[nt].v = __builtin_amdgcn_wmma_f32_16x16x32_bf16(
                            false, pa.v, false, vb.v, (short)0, o[nt].v, false, false);
            }
        }
    }

    // normalize and store to [B, L, C] bf16 for the output projection
#pragma unroll
    for (int nt = 0; nt < 4; ++nt)
#pragma unroll
        for (int i = 0; i < 8; ++i) {
            int row = qrow0 + i + 8 * half;
            float inv = (lrow[i] > 0.0f) ? (1.0f / lrow[i]) : 0.0f;
            float val = o[nt].f[i] * inv;
            obf[((size_t)b_ * LL + row) * CC + h * DD + nt * 16 + r] = f2bf(val);
        }
}

// ---------------------------------------------------------------------------
// Kernel 5: out = o @ Wproj + bproj  (M=4096, N=512, K=512), f32 out.
// ---------------------------------------------------------------------------
__global__ void k_out_gemm(const unsigned short* __restrict__ obf,
                           const float* __restrict__ Wproj,
                           const float* __restrict__ bproj,
                           float* __restrict__ out) {
    const int gw   = blockIdx.x * 8 + (threadIdx.x >> 5);
    const int lane = threadIdx.x & 31;
    const int half = lane >> 4, r = lane & 15;
    const int tM = gw / 32, tN = gw % 32;

    F32x8 c;
#pragma unroll
    for (int i = 0; i < 8; ++i) c.f[i] = 0.0f;

    for (int k = 0; k < CC; k += 32) {
        if (k + 32 < CC) {
            __builtin_prefetch(&Wproj[(size_t)(k + 32 + 16 * half) * CC + tN * 16 + r], 0, 1);
        }
        BF16x16 a, b;
#pragma unroll
        for (int e = 0; e < 16; ++e) {
            a.u[e] = obf[(size_t)(tM * 16 + r) * CC + k + a_kidx(e, half)];
            b.u[e] = f2bf(Wproj[(size_t)(k + a_kidx(e, half)) * CC + tN * 16 + r]);
        }
        c.v = __builtin_amdgcn_wmma_f32_16x16x32_bf16(false, a.v, false, b.v,
                                                      (short)0, c.v, false, false);
    }
#pragma unroll
    for (int i = 0; i < 8; ++i) {
        int m = tM * 16 + i + 8 * half;
        int n = tN * 16 + r;
        out[(size_t)m * CC + n] = c.f[i] + bproj[n];
    }
}

// ---------------------------------------------------------------------------
// Host launcher
// ---------------------------------------------------------------------------
extern "C" void kernel_launch(void* const* d_in, const int* in_sizes, int n_in,
                              void* d_out, int out_size, void* d_ws, size_t ws_size,
                              hipStream_t stream) {
    const float* x     = (const float*)d_in[0];
    const float* Wqkv  = (const float*)d_in[1];
    const float* bqkv  = (const float*)d_in[2];
    const float* Wproj = (const float*)d_in[3];
    const float* bproj = (const float*)d_in[4];
    float* out = (float*)d_out;

    const size_t nBHLD = (size_t)BB * HH * LL * DD;   // 2,097,152

    // workspace carve-up (~32.3 MB total)
    char* ws = (char*)d_ws;
    float* qf = (float*)ws;                         ws += nBHLD * sizeof(float);
    float* kf = (float*)ws;                         ws += nBHLD * sizeof(float);
    unsigned short* qbf = (unsigned short*)ws;      ws += nBHLD * sizeof(unsigned short);
    unsigned short* kbf = (unsigned short*)ws;      ws += nBHLD * sizeof(unsigned short);
    unsigned short* vbf = (unsigned short*)ws;      ws += nBHLD * sizeof(unsigned short);
    unsigned short* obf = (unsigned short*)ws;      ws += nBHLD * sizeof(unsigned short);
    float* qm   = (float*)ws;                       ws += (size_t)BB * HH * NQ * DD * sizeof(float);
    float* km   = (float*)ws;                       ws += (size_t)BB * HH * NK * DD * sizeof(float);
    float* qsim = (float*)ws;                       ws += (size_t)BB * HH * NQ * sizeof(float);
    float* ksim = (float*)ws;                       ws += (size_t)BB * HH * NK * sizeof(float);
    int*   mask = (int*)ws;

    // 1) QKV GEMM: 256x96 tiles, 8 waves/block -> 3072 blocks
    k_qkv_gemm<<<3072, 256, 0, stream>>>(x, Wqkv, bqkv, qf, kf, qbf, kbf, vbf);

    // 2) block metadata for Q (blk=128, nb=16) and K (blk=64, nb=32)
    k_block_meta<<<BB * HH * NQ, 128, 0, stream>>>(qf, 128, NQ, qm, qsim);
    k_block_meta<<<BB * HH * NK, 128, 0, stream>>>(kf, 64, NK, km, ksim);

    // 3) pooled softmax + CDF keep mask: one wave per (b,h,qblock)
    k_pool_mask<<<BB * HH * NQ, 32, 0, stream>>>(qm, km, qsim, ksim, mask);

    // 4) block-sparse flash attention (async LDS staging + WMMA)
    k_sparse_attn<<<BB * HH * NQ, 256, 0, stream>>>(qbf, kbf, vbf, mask, obf);

    // 5) output projection: 256x32 tiles, 8 waves/block -> 1024 blocks
    k_out_gemm<<<1024, 256, 0, stream>>>(obf, Wproj, bproj, out);
}
